// Model_5274219840279
// MI455X (gfx1250) — matrix operand order repaired and checked by reading, not compiled, not run on hardware
//
#include <hip/hip_runtime.h>
#include <hip/hip_bf16.h>
#include <math.h>

typedef __attribute__((ext_vector_type(16))) __bf16 v16bf;
typedef __attribute__((ext_vector_type(8)))  float  v8f;

__device__ __forceinline__ float relu_(float x) { return x > 0.f ? x : 0.f; }

#define HAVE_TDM (__has_builtin(__builtin_amdgcn_tensor_load_to_lds) && \
                  __has_builtin(__builtin_amdgcn_s_wait_tensorcnt))

// ===========================================================================
// Weight repack: float OIHW -> bf16 tiles in a contiguous-k WMMA A-operand lane
// layout (16-bit A 16x32; B uses the same k order):  lane l holds M=l&15; element j of
// v16bf holds K = j + 16*(l>=16).
// Output layout: [coTile][kChunk][lane(32)][16] contiguous bf16.
// ===========================================================================
__global__ void repack_conv_w_k(const float* __restrict__ w, __bf16* __restrict__ wp,
                                int Cout, int Ktot, int coTiles)
{
    int kChunks = (Ktot + 31) >> 5;
    int total = coTiles * kChunks * 512;
    int i = blockIdx.x * 256 + threadIdx.x;
    if (i >= total) return;
    int j = i & 15;
    int lane = (i >> 4) & 31;
    int tile = i >> 9;                 // mt*kChunks + kc
    int kc = tile % kChunks;
    int mt = tile / kChunks;
    int laneLo = lane & 15, hi = lane >> 4;
    int v = j >> 1, p = j & 1;
    int kA = j + (hi << 4);
    int kk = kc * 32 + kA;
    int co = mt * 16 + laneLo;
    float val = (kk < Ktot && co < Cout) ? w[(long)co * Ktot + kk] : 0.f;
    wp[i] = (__bf16)val;
}

// ConvTranspose2d weights [Cin][Cout][4][4] -> per tap-parity class
// (class = ky0*2+kx0), K index = cin*4 + ty*2 + tx, same A lane layout.
// Output: [class(4)][coTile][kChunk][lane][16].
__global__ void repack_convT_w_k(const float* __restrict__ w, __bf16* __restrict__ wp,
                                 int Cin, int Cout, int coTiles)
{
    int Ktot = Cin << 2;
    int kChunks = Ktot >> 5;
    int total = 4 * coTiles * kChunks * 512;
    int i = blockIdx.x * 256 + threadIdx.x;
    if (i >= total) return;
    int j = i & 15;
    int lane = (i >> 4) & 31;
    int tile = i >> 9;
    int kc = tile % kChunks; tile /= kChunks;
    int mt = tile % coTiles;
    int cls = tile / coTiles;
    int ky0 = cls >> 1, kx0 = cls & 1;
    int laneLo = lane & 15, hi = lane >> 4;
    int v = j >> 1, p = j & 1;
    int kA = j + (hi << 4);
    int kk = kc * 32 + kA;
    int co = mt * 16 + laneLo;
    float val = 0.f;
    if (kk < Ktot && co < Cout) {
        int cin = kk >> 2, tap = kk & 3;
        int ky = ky0 + ((tap >> 1) << 1);
        int kx = kx0 + ((tap & 1) << 1);
        val = w[((long)(cin * Cout + co) * 4 + ky) * 4 + kx];
    }
    wp[i] = (__bf16)val;
}

// ===========================================================================
// NCHW conv as implicit GEMM, one wave per block, M-register-blocked.
// ===========================================================================
template <int MT>
__global__ __launch_bounds__(32)
void conv2d_wmma_k(const float* __restrict__ in, const __bf16* __restrict__ wp,
                   const float* __restrict__ bias, const float* __restrict__ res,
                   float* __restrict__ out,
                   int N, int Cin, int H, int W,
                   int Cout, int OH, int OW,
                   int kh, int kw, int stride, int pad,
                   int inRelu, int outRelu)
{
    const int lane = threadIdx.x;
    const int laneLo = lane & 15;
    const int hi = lane >> 4;

    const int tilesPerRow = OW >> 4;
    const int t = blockIdx.x;
    const int oxBase = (t % tilesPerRow) << 4;
    const int row = t / tilesPerRow;
    const int oy = row % OH;
    const int n = row / OH;

    const int khw = kh * kw;
    const int Ktot = Cin * khw;
    const int kChunks = (Ktot + 31) >> 5;
    const int ox = oxBase + laneLo;
    const long laneSlot = (long)lane * 16;

    v8f acc[MT];
#pragma unroll
    for (int m = 0; m < MT; ++m) acc[m] = (v8f){};

    for (int kb = 0, kc = 0; kb < Ktot; kb += 32, ++kc) {
        v16bf b;
#pragma unroll
        for (int j = 0; j < 16; ++j) {
            int kk = kb + (hi << 4) + j;
            int cin = kk / khw;
            int rem = kk - cin * khw;
            int ky = rem / kw;
            int kx = rem - ky * kw;
            int iy = oy * stride - pad + ky;
            int ix = ox * stride - pad + kx;
            bool valid = (kk < Ktot) & ((unsigned)iy < (unsigned)H) &
                         ((unsigned)ix < (unsigned)W);
            long addr = valid ? (((long)(n * Cin + cin) * H + iy) * W + ix) : 0;
            float xv = in[addr];
            xv = valid ? xv : 0.f;
            if (inRelu) xv = relu_(xv);
            b[j] = (__bf16)xv;
        }
#pragma unroll
        for (int m = 0; m < MT; ++m) {
            long mtg = (long)blockIdx.y * MT + m;
            const __bf16* ap = wp + ((mtg * kChunks + kc) << 9) + laneSlot;
            if (m == 0 && kb + 32 < Ktot) __builtin_prefetch(ap + 512, 0, 0);
            v16bf a = *reinterpret_cast<const v16bf*>(ap);
            acc[m] = __builtin_amdgcn_wmma_f32_16x16x32_bf16(false, a, false, b,
                                                             (short)0, acc[m],
                                                             false, false);
        }
    }

#pragma unroll
    for (int m = 0; m < MT; ++m) {
        int coBase = (blockIdx.y * MT + m) << 4;
#pragma unroll
        for (int r = 0; r < 8; ++r) {
            int co = coBase + r + (hi << 3);
            if (co < Cout) {
                float val = acc[m][r];
                if (bias) val += bias[co];
                long idx = ((long)(n * Cout + co) * OH + oy) * OW + ox;
                if (res) val += res[idx];
                if (outRelu) val = relu_(val);
                out[idx] = val;
            }
        }
    }
}

// ===========================================================================
// Transposed conv, stride=2, k=4, pad=1 (gather form), M-register-blocked.
// ===========================================================================
template <int MT>
__global__ __launch_bounds__(32)
void convT_wmma_k(const float* __restrict__ in, const __bf16* __restrict__ wp,
                  const float* __restrict__ bias, float* __restrict__ out,
                  int N, int Cin, int H, int W, int Cout, int OH, int OW,
                  int outRelu)
{
    const int lane = threadIdx.x;
    const int laneLo = lane & 15;
    const int hi = lane >> 4;

    const int tilesPerRow = OW >> 4;
    const int half = tilesPerRow >> 1;
    const int t = blockIdx.x;
    int t2 = t % tilesPerRow;
    const int row = t / tilesPerRow;
    const int oy = row % OH;
    const int n = row / OH;
    int oxBase = (t2 < half) ? (t2 << 5) : (((t2 - half) << 5) + 1);
    const int ox = oxBase + (laneLo << 1);
    const int kx0 = (ox + 1) & 1;
    const int ky0 = (oy + 1) & 1;
    const int cls = (ky0 << 1) + kx0;
    const int coTiles = gridDim.y * MT;
    const int Ktot = Cin << 2;
    const int kChunks = Ktot >> 5;
    const long laneSlot = (long)lane * 16;

    v8f acc[MT];
#pragma unroll
    for (int m = 0; m < MT; ++m) acc[m] = (v8f){};

    for (int kb = 0, kc = 0; kb < Ktot; kb += 32, ++kc) {
        v16bf b;
#pragma unroll
        for (int j = 0; j < 16; ++j) {
            int kk = kb + (hi << 4) + j;
            int cin = kk >> 2, tap = kk & 3;
            int ky = ky0 + ((tap >> 1) << 1);
            int kx = kx0 + ((tap & 1) << 1);
            int tyn = oy + 1 - ky;
            int txn = ox + 1 - kx;
            int iy = tyn >> 1, ix = txn >> 1;
            bool valid = (kk < Ktot) & (tyn >= 0) & (txn >= 0) & (iy < H) & (ix < W);
            long addr = valid ? (((long)(n * Cin + cin) * H + iy) * W + ix) : 0;
            float xv = in[addr];
            xv = valid ? xv : 0.f;
            b[j] = (__bf16)xv;
        }
#pragma unroll
        for (int m = 0; m < MT; ++m) {
            long mtg = (long)blockIdx.y * MT + m;
            const __bf16* ap = wp + ((((long)cls * coTiles + mtg) * kChunks + kc) << 9)
                             + laneSlot;
            if (m == 0 && kb + 32 < Ktot) __builtin_prefetch(ap + 512, 0, 0);
            v16bf a = *reinterpret_cast<const v16bf*>(ap);
            acc[m] = __builtin_amdgcn_wmma_f32_16x16x32_bf16(false, a, false, b,
                                                             (short)0, acc[m],
                                                             false, false);
        }
    }

#pragma unroll
    for (int m = 0; m < MT; ++m) {
        int coBase = (blockIdx.y * MT + m) << 4;
#pragma unroll
        for (int r = 0; r < 8; ++r) {
            int co = coBase + r + (hi << 3);
            if (co < Cout) {
                float val = acc[m][r] + (bias ? bias[co] : 0.f);
                if (outRelu) val = relu_(val);
                out[((long)(n * Cout + co) * OH + oy) * OW + ox] = val;
            }
        }
    }
}

// ===========================================================================
__global__ void addnoise_k(const float* __restrict__ zc, const float* __restrict__ noise,
                           float* __restrict__ outZ, float* __restrict__ znh)
{
    int i = blockIdx.x * 256 + threadIdx.x;
    if (i >= 4194304) return;
    float v = zc[i] + 10.f * noise[i];
    outZ[i] = v;
    int d = (i >> 12) & 63;
    int n = i >> 18;
    int yx = i & 4095;
    znh[((long)((n << 12) + yx)) * 64 + d] = v;
}

__global__ void enorm_k(const float* __restrict__ emb, float* __restrict__ e2)
{
    int k = blockIdx.x * blockDim.x + threadIdx.x;
    if (k < 512) {
        float s = 0.f;
        for (int d = 0; d < 64; ++d) { float v = emb[k * 64 + d]; s += v * v; }
        e2[k] = s;
    }
}

__global__ void vqinit_k(float* cnt, float* lacc)
{
    int i = blockIdx.x * blockDim.x + threadIdx.x;
    if (i < 512) cnt[i] = 0.f;
    if (i == 512) lacc[0] = 0.f;
}

// ===========================================================================
// VQ argmin via WMMA. Codebook staged into LDS by the Tensor Data Mover
// (two 64KB tiles, D# per ISA 08_async_tensor §8), synchronized with
// TENSORcnt; B operands come from LDS (ds_load_b128). The descriptor's
// lds_addr is derived from the shared array's flat address (low 32 bits =
// LDS byte offset), which also makes the array escape so the side-effecting
// TDM intrinsic is assumed to write it (keeps the LDS reads alive).
// Min-reduction is branchless: 64-bit order-preserving (value,index) keys
// min-reduced with __shfl_xor ternaries (v_min_u64 + ds_bpermute).
// ===========================================================================
__device__ __forceinline__ unsigned long long fkey_(float v, int idx)
{
    unsigned ub = __float_as_uint(v);
    ub = ((int)ub < 0) ? ~ub : (ub | 0x80000000u);
    return ((unsigned long long)ub << 32) | (unsigned)idx;
}

__global__ __launch_bounds__(32)
void vq_argmin_k(const float* __restrict__ znh, const float* __restrict__ emb,
                 const float* __restrict__ e2, int* __restrict__ idxOut)
{
    const int lane = threadIdx.x;
    const int laneLo = lane & 15;
    const int hi = lane >> 4;
    const int pixBase = blockIdx.x << 4;
    const float* za = znh + ((long)(pixBase + laneLo)) * 64;

    // Preload A (16 pixels x K=64) once: 2 chunks of v16bf per lane.
    v16bf aChunk[2];
#pragma unroll
    for (int kc = 0; kc < 2; ++kc) {
        int kb = kc * 32;
        const float4* pa0 = reinterpret_cast<const float4*>(za + kb + (hi << 4));
        const float4* pa1 = reinterpret_cast<const float4*>(za + kb + (hi << 4) + 8);
        float4 f0 = pa0[0], f1 = pa0[1], f2 = pa1[0], f3 = pa1[1];
        v16bf a;
        a[0] = (__bf16)f0.x;  a[1] = (__bf16)f0.y;  a[2] = (__bf16)f0.z;  a[3] = (__bf16)f0.w;
        a[4] = (__bf16)f1.x;  a[5] = (__bf16)f1.y;  a[6] = (__bf16)f1.z;  a[7] = (__bf16)f1.w;
        a[8] = (__bf16)f2.x;  a[9] = (__bf16)f2.y;  a[10] = (__bf16)f2.z; a[11] = (__bf16)f2.w;
        a[12] = (__bf16)f3.x; a[13] = (__bf16)f3.y; a[14] = (__bf16)f3.z; a[15] = (__bf16)f3.w;
        aChunk[kc] = a;
    }

    unsigned long long best[8];
#pragma unroll
    for (int r = 0; r < 8; ++r) best[r] = ~0ull;

    auto do_tile = [&](int cb, const float* bbase) {
        v8f c = {};
#pragma unroll
        for (int kc = 0; kc < 2; ++kc) {
            const float4* pb = reinterpret_cast<const float4*>(bbase + kc * 32);
            float4 g0 = pb[0], g1 = pb[1], g2 = pb[2], g3 = pb[3];
            v16bf b;
            b[0] = (__bf16)g0.x;  b[1] = (__bf16)g0.y;  b[2] = (__bf16)g0.z;  b[3] = (__bf16)g0.w;
            b[4] = (__bf16)g1.x;  b[5] = (__bf16)g1.y;  b[6] = (__bf16)g1.z;  b[7] = (__bf16)g1.w;
            b[8] = (__bf16)g2.x;  b[9] = (__bf16)g2.y;  b[10] = (__bf16)g2.z; b[11] = (__bf16)g2.w;
            b[12] = (__bf16)g3.x; b[13] = (__bf16)g3.y; b[14] = (__bf16)g3.z; b[15] = (__bf16)g3.w;
            c = __builtin_amdgcn_wmma_f32_16x16x32_bf16(false, aChunk[kc], false, b,
                                                        (short)0, c, false, false);
        }
        float en = e2[cb + laneLo];
#pragma unroll
        for (int r = 0; r < 8; ++r) {
            unsigned long long key = fkey_(en - 2.0f * c[r], cb + laneLo);
#pragma unroll
            for (int off = 1; off < 16; off <<= 1) {
                unsigned long long ok = __shfl_xor(key, off, 32);
                key = (ok < key) ? ok : key;
            }
            best[r] = (key < best[r]) ? key : best[r];
        }
    };

#if HAVE_TDM
    typedef __attribute__((ext_vector_type(4))) unsigned int u32x4;
    typedef __attribute__((ext_vector_type(8))) int  i32x8;
    typedef __attribute__((ext_vector_type(4))) int  i32x4;
    __shared__ float ecb[256 * 64];                // 64 KB codebook stage
    // Flat LDS addresses carry the LDS byte offset in the low 32 bits; the
    // ptrtoint also makes `ecb` escape (TDM intrinsic may write it).
    unsigned ldsBase = (unsigned)(unsigned long long)(void*)ecb;
    for (int st = 0; st < 2; ++st) {
        if (st) asm volatile("s_wait_dscnt 0" ::: "memory");   // reads done before overwrite
        unsigned long long ga = (unsigned long long)(emb + (long)st * 256 * 64);
        u32x4 g0;
        g0[0] = 1u;                                            // count=1 valid user desc
        g0[1] = ldsBase;                                       // lds_addr (bytes)
        g0[2] = (unsigned)(ga & 0xffffffffu);                  // global_addr[31:0]
        g0[3] = (unsigned)((ga >> 32) & 0x1ffffffu) | (2u << 30); // addr[56:32] | type=2
        i32x8 g1;
        g1[0] = (int)(2u << 16);        // data_size = 4B
        g1[1] = (int)(64u << 16);       // tensor_dim0 = 64 (bits 79:48 lo)
        g1[2] = (int)(512u << 16);      // dim0 hi=0 | tensor_dim1 = 512 (bits 111:80 lo)
        g1[3] = (int)(64u << 16);       // dim1 hi=0 | tile_dim0 = 64 (bits 127:112)
        g1[4] = 256;                    // tile_dim1 = 256 rows
        g1[5] = 64;                     // tensor_dim0_stride = 64 (bits 207:160 lo)
        g1[6] = 0;
        g1[7] = 0;
        i32x4 gz4 = {0, 0, 0, 0};       // groups 2/3 unused (2D tensor)
        i32x8 gz8 = {0, 0, 0, 0, 0, 0, 0, 0};
        __builtin_amdgcn_tensor_load_to_lds(g0, g1, gz4, gz4, gz8, 0);
        __builtin_amdgcn_s_wait_tensorcnt((short)0);
        asm volatile("" ::: "memory");  // don't hoist LDS reads above the wait
        for (int cbl = 0; cbl < 256; cbl += 16)
            do_tile(st * 256 + cbl, ecb + (long)(cbl + laneLo) * 64 + (hi << 4));
    }
#else
    for (int cb = 0; cb < 512; cb += 16)
        do_tile(cb, emb + (long)(cb + laneLo) * 64 + (hi << 4));
#endif

    if (laneLo == 0) {
#pragma unroll
        for (int r = 0; r < 8; ++r)
            idxOut[pixBase + (hi << 3) + r] = (int)(best[r] & 0xffffffffu);
    }
}

// ===========================================================================
__global__ void vq_gather_k(const float* __restrict__ znh, const float* __restrict__ emb,
                            const int* __restrict__ idx, float* __restrict__ q,
                            float* cnt, float* lacc)
{
    __shared__ float sm[256];
    int pix = blockIdx.x * 256 + threadIdx.x;
    int n = pix >> 12, y = (pix >> 6) & 63, x = pix & 63;
    int id = idx[pix];
    float s = 0.f;
    for (int d = 0; d < 64; ++d) {
        float qv = emb[(long)id * 64 + d];
        float zv = znh[(long)pix * 64 + d];
        q[(((long)(n * 64 + d)) << 12) + (y << 6) + x] = qv;
        float df = qv - zv;
        s += df * df;
    }
    atomicAdd(&cnt[id], 1.0f);
    sm[threadIdx.x] = s;
    __syncthreads();
    for (int off = 128; off > 0; off >>= 1) {
        if (threadIdx.x < off) sm[threadIdx.x] += sm[threadIdx.x + off];
        __syncthreads();
    }
    if (threadIdx.x == 0) atomicAdd(lacc, sm[0]);
}

__global__ void vq_final_k(const float* __restrict__ cnt, const float* __restrict__ lacc,
                           float* outLoss, float* outPerp)
{
    __shared__ float sm[512];
    int t = threadIdx.x;
    float p = cnt[t] * (1.0f / 65536.0f);
    sm[t] = p * logf(p + 1e-10f);
    __syncthreads();
    for (int off = 256; off > 0; off >>= 1) {
        if (t < off) sm[t] += sm[t + off];
        __syncthreads();
    }
    if (t == 0) {
        outPerp[0] = expf(-sm[0]);
        outLoss[0] = 0.25f * lacc[0] * (1.0f / (65536.0f * 64.0f));
    }
}

// ===========================================================================
extern "C" void kernel_launch(void* const* d_in, const int* in_sizes, int n_in,
                              void* d_out, int out_size, void* d_ws, size_t ws_size,
                              hipStream_t stream)
{
    (void)in_sizes; (void)n_in; (void)out_size; (void)ws_size;
    const float* x     = (const float*)d_in[0];
    const float* noise = (const float*)d_in[1];
    const float* e_w1  = (const float*)d_in[2];
    const float* e_b1  = (const float*)d_in[3];
    const float* e_w2  = (const float*)d_in[4];
    const float* e_b2  = (const float*)d_in[5];
    const float* e_w3  = (const float*)d_in[6];
    const float* e_b3  = (const float*)d_in[7];
    const float* er0w3 = (const float*)d_in[8];
    const float* er0w1 = (const float*)d_in[9];
    const float* er1w3 = (const float*)d_in[10];
    const float* er1w1 = (const float*)d_in[11];
    const float* pv_w  = (const float*)d_in[12];
    const float* pv_b  = (const float*)d_in[13];
    const float* emb   = (const float*)d_in[14];
    const float* d_w1  = (const float*)d_in[15];
    const float* d_b1  = (const float*)d_in[16];
    const float* dr0w3 = (const float*)d_in[17];
    const float* dr0w1 = (const float*)d_in[18];
    const float* dr1w3 = (const float*)d_in[19];
    const float* dr1w1 = (const float*)d_in[20];
    const float* dt_w1 = (const float*)d_in[21];
    const float* dt_b1 = (const float*)d_in[22];
    const float* dt_w2 = (const float*)d_in[23];
    const float* dt_b2 = (const float*)d_in[24];

    float* ws = (float*)d_ws;
    float* A    = ws;                   // 16,777,216  [16,64,128,128]
    float* B    = ws + 16777216;        //  8,388,608  [16,128,64,64]
    float* C    = ws + 25165824;        //  8,388,608  [16,128,64,64]
    float* T    = ws + 33554432;        //  2,097,152  [16,32,64,64]
    float* Z    = ws + 35651584;        //  4,194,304  [16,64,64,64] (pv out)
    float* Q    = ws + 39845888;        //  4,194,304  [16,64,64,64]
    float* ZNH  = ws + 44040192;        //  4,194,304  [65536,64]
    float* E2   = ws + 48234496;        //  512
    float* CNT  = ws + 48235008;        //  512
    float* LACC = ws + 48235520;        //  1 (+3 pad)
    int*   IDX  = (int*)(ws + 48235524);    // 65536
    __bf16* WP  = (__bf16*)(ws + 48301064); // packed weights (32B aligned)

    __bf16* wp_e1  = WP + 0;
    __bf16* wp_e2  = WP + 2048;
    __bf16* wp_e3  = WP + 133120;
    __bf16* wp_r0a = WP + 280576;
    __bf16* wp_r0b = WP + 317440;
    __bf16* wp_r1a = WP + 321536;
    __bf16* wp_r1b = WP + 358400;
    __bf16* wp_pv  = WP + 362496;
    __bf16* wp_d1  = WP + 370688;
    __bf16* wp_s0a = WP + 444416;
    __bf16* wp_s0b = WP + 481280;
    __bf16* wp_s1a = WP + 485376;
    __bf16* wp_s1b = WP + 522240;
    __bf16* wp_t1  = WP + 526336;
    __bf16* wp_t2  = WP + 657408;

    float* out     = (float*)d_out;
    float* outLoss = out;
    float* outRec  = out + 1;
    float* outPerp = out + 3145729;
    float* outZ    = out + 3145730;

    dim3 wv(32), tb(256);

    // --- Weight repack ---
    repack_conv_w_k<<<dim3(8),   tb, 0, stream>>>(e_w1,  wp_e1,  64,  16,   4);
    repack_conv_w_k<<<dim3(512), tb, 0, stream>>>(e_w2,  wp_e2,  128, 1024, 8);
    repack_conv_w_k<<<dim3(576), tb, 0, stream>>>(e_w3,  wp_e3,  128, 1152, 8);
    repack_conv_w_k<<<dim3(144), tb, 0, stream>>>(er0w3, wp_r0a, 32,  1152, 2);
    repack_conv_w_k<<<dim3(16),  tb, 0, stream>>>(er0w1, wp_r0b, 128, 32,   8);
    repack_conv_w_k<<<dim3(144), tb, 0, stream>>>(er1w3, wp_r1a, 32,  1152, 2);
    repack_conv_w_k<<<dim3(16),  tb, 0, stream>>>(er1w1, wp_r1b, 128, 32,   8);
    repack_conv_w_k<<<dim3(32),  tb, 0, stream>>>(pv_w,  wp_pv,  64,  128,  4);
    repack_conv_w_k<<<dim3(288), tb, 0, stream>>>(d_w1,  wp_d1,  128, 576,  8);
    repack_conv_w_k<<<dim3(144), tb, 0, stream>>>(dr0w3, wp_s0a, 32,  1152, 2);
    repack_conv_w_k<<<dim3(16),  tb, 0, stream>>>(dr0w1, wp_s0b, 128, 32,   8);
    repack_conv_w_k<<<dim3(144), tb, 0, stream>>>(dr1w3, wp_s1a, 32,  1152, 2);
    repack_conv_w_k<<<dim3(16),  tb, 0, stream>>>(dr1w1, wp_s1b, 128, 32,   8);
    repack_convT_w_k<<<dim3(512), tb, 0, stream>>>(dt_w1, wp_t1, 128, 64, 4);
    repack_convT_w_k<<<dim3(64),  tb, 0, stream>>>(dt_w2, wp_t2, 64,  3,  1);

    // --- Encoder ---
    conv2d_wmma_k<4><<<dim3(16*128*8, 1), wv, 0, stream>>>(x, wp_e1, e_b1, nullptr, A,
        16, 1, 256, 256, 64, 128, 128, 4, 4, 2, 1, 0, 1);
    conv2d_wmma_k<4><<<dim3(16*64*4, 2), wv, 0, stream>>>(A, wp_e2, e_b2, nullptr, B,
        16, 64, 128, 128, 128, 64, 64, 4, 4, 2, 1, 0, 1);
    conv2d_wmma_k<4><<<dim3(16*64*4, 2), wv, 0, stream>>>(B, wp_e3, e_b3, nullptr, C,
        16, 128, 64, 64, 128, 64, 64, 3, 3, 1, 1, 0, 0);
    conv2d_wmma_k<2><<<dim3(16*64*4, 1), wv, 0, stream>>>(C, wp_r0a, nullptr, nullptr, T,
        16, 128, 64, 64, 32, 64, 64, 3, 3, 1, 1, 1, 0);
    conv2d_wmma_k<4><<<dim3(16*64*4, 2), wv, 0, stream>>>(T, wp_r0b, nullptr, C, B,
        16, 32, 64, 64, 128, 64, 64, 1, 1, 1, 0, 1, 0);
    conv2d_wmma_k<2><<<dim3(16*64*4, 1), wv, 0, stream>>>(B, wp_r1a, nullptr, nullptr, T,
        16, 128, 64, 64, 32, 64, 64, 3, 3, 1, 1, 1, 0);
    conv2d_wmma_k<4><<<dim3(16*64*4, 2), wv, 0, stream>>>(T, wp_r1b, nullptr, B, C,
        16, 32, 64, 64, 128, 64, 64, 1, 1, 1, 0, 1, 1);
    conv2d_wmma_k<4><<<dim3(16*64*4, 1), wv, 0, stream>>>(C, wp_pv, pv_b, nullptr, Z,
        16, 128, 64, 64, 64, 64, 64, 1, 1, 1, 0, 0, 0);
    addnoise_k<<<dim3(16384), tb, 0, stream>>>(Z, noise, outZ, ZNH);

    // --- VQ ---
    vqinit_k<<<dim3(3), tb, 0, stream>>>(CNT, LACC);
    enorm_k<<<dim3(2), tb, 0, stream>>>(emb, E2);
    vq_argmin_k<<<dim3(4096), wv, 0, stream>>>(ZNH, emb, E2, IDX);
    vq_gather_k<<<dim3(256), tb, 0, stream>>>(ZNH, emb, IDX, Q, CNT, LACC);
    vq_final_k<<<dim3(1), dim3(512), 0, stream>>>(CNT, LACC, outLoss, outPerp);

    // --- Decoder ---
    conv2d_wmma_k<4><<<dim3(16*64*4, 2), wv, 0, stream>>>(Q, wp_d1, d_b1, nullptr, B,
        16, 64, 64, 64, 128, 64, 64, 3, 3, 1, 1, 0, 0);
    conv2d_wmma_k<2><<<dim3(16*64*4, 1), wv, 0, stream>>>(B, wp_s0a, nullptr, nullptr, T,
        16, 128, 64, 64, 32, 64, 64, 3, 3, 1, 1, 1, 0);
    conv2d_wmma_k<4><<<dim3(16*64*4, 2), wv, 0, stream>>>(T, wp_s0b, nullptr, B, C,
        16, 32, 64, 64, 128, 64, 64, 1, 1, 1, 0, 1, 0);
    conv2d_wmma_k<2><<<dim3(16*64*4, 1), wv, 0, stream>>>(C, wp_s1a, nullptr, nullptr, T,
        16, 128, 64, 64, 32, 64, 64, 3, 3, 1, 1, 1, 0);
    conv2d_wmma_k<4><<<dim3(16*64*4, 2), wv, 0, stream>>>(T, wp_s1b, nullptr, C, B,
        16, 32, 64, 64, 128, 64, 64, 1, 1, 1, 0, 1, 1);
    convT_wmma_k<4><<<dim3(16*128*8, 1), wv, 0, stream>>>(B, wp_t1, dt_b1, A,
        16, 128, 64, 64, 64, 128, 128, 1);
    convT_wmma_k<1><<<dim3(16*256*16, 1), wv, 0, stream>>>(A, wp_t2, dt_b2, outRec,
        16, 64, 128, 128, 3, 256, 256, 0);
}
